// Model_7627861917785
// MI455X (gfx1250) — matrix operand order repaired and checked by reading, not compiled, not run on hardware
//
#include <hip/hip_runtime.h>
#include <hip/hip_bf16.h>
#include <math.h>

#define BATCH   16
#define L_CYC   20
#define CDL     128
#define SEQQ    2560          // L_CYC*CDL
#define DMODEL  512
#define DFF     2048
#define ENCIN   3
#define TOPK    7
#define ELAYERS 2

typedef __attribute__((ext_vector_type(16))) __bf16 v16bf;
typedef __attribute__((ext_vector_type(8)))  float  v8f;

union Frag {
    uint4  q2[2];   // two 16-byte halves (VGPRs 0-3, 4-7)
    v16bf  v;
};

__device__ __forceinline__ unsigned short f2bf(float f) {
    unsigned u = __float_as_uint(f);
    u = (u + 0x7FFFu + ((u >> 16) & 1u)) >> 16;   // round-to-nearest-even
    return (unsigned short)u;
}

__device__ __forceinline__ float gelu_f(float x) {
    return 0.5f * x * (1.0f + erff(x * 0.70710678118654752f));
}

// ---------------------------------------------------------------- embed
// h[b,s,o] = sum_{c,kw} x[b,(s+kw-1)%S,c] * Wemb[o,c,kw],  x = masked data
__global__ void embed_kernel(const float* __restrict__ data,
                             const float* __restrict__ mask,
                             const float* __restrict__ Wemb,
                             float* __restrict__ h) {
    size_t idx = (size_t)blockIdx.x * blockDim.x + threadIdx.x;
    if (idx >= (size_t)BATCH * SEQQ * DMODEL) return;
    int o = (int)(idx % DMODEL);
    size_t bs = idx / DMODEL;
    int s = (int)(bs % SEQQ);
    int b = (int)(bs / SEQQ);
    float acc = 0.f;
#pragma unroll
    for (int kw = 0; kw < 3; ++kw) {
        int sp = s + kw - 1;
        if (sp < 0) sp += SEQQ;
        if (sp >= SEQQ) sp -= SEQQ;
        int l = sp >> 7, t = sp & 127;
        float m = mask[b * L_CYC + l];
#pragma unroll
        for (int c = 0; c < ENCIN; ++c) {
            float xv = data[(((size_t)b * L_CYC + l) * ENCIN + c) * CDL + t] * m;
            acc += xv * Wemb[(o * ENCIN + c) * 3 + kw];
        }
    }
    h[idx] = acc;
}

// ---------------------------------------------------------------- converts
__global__ void f2bf_kernel(const float* __restrict__ x,
                            unsigned short* __restrict__ y, size_t n) {
    size_t i = (size_t)blockIdx.x * blockDim.x + threadIdx.x;
    if (i < n) y[i] = f2bf(x[i]);
}

// W is KxN fp32 row-major; Wt is NxK bf16 row-major
__global__ void wtT_kernel(const float* __restrict__ W,
                           unsigned short* __restrict__ Wt, int K, int N) {
    size_t i = (size_t)blockIdx.x * blockDim.x + threadIdx.x;
    if (i >= (size_t)K * N) return;
    int n = (int)(i / K), k = (int)(i % K);
    Wt[i] = f2bf(W[(size_t)k * N + n]);
}

// ---------------------------------------------------------------- WMMA GEMM
// Each wave computes a 32x64 output block: 2 M-frags x 4 N-frags (8 acc).
// C = op(A[MxK,bf16] * Bt[NxK,bf16] [+bias] [gelu] [+resid])
// Output: fp32 to C, or bf16 to Cb (if Cb != null).
__global__ void wmma_gemm_kernel(const unsigned short* __restrict__ A,
                                 const unsigned short* __restrict__ Bt,
                                 const float* __restrict__ bias,
                                 const float* __restrict__ resid,
                                 float* __restrict__ C,
                                 unsigned short* __restrict__ Cb,
                                 int M, int N, int K, int act) {
    int wid  = blockIdx.x * (blockDim.x >> 5) + (threadIdx.x >> 5);
    int lane = threadIdx.x & 31;
    int nG = N >> 6;                       // 64-wide col groups
    int tm = wid / nG, tn = wid % nG;
    if (tm >= (M >> 5)) return;
    int m0 = tm << 5, n0 = tn << 6;
    int lm = lane & 15, lh = lane >> 4;

    const unsigned short* Arow0 = A + (size_t)(m0 + lm) * K + lh * 8;
    const unsigned short* Arow1 = Arow0 + (size_t)16 * K;
    const unsigned short* Brow  = Bt + (size_t)(n0 + lm) * K + lh * 8;

    v8f acc[2][4] = {};
    for (int k0 = 0; k0 < K; k0 += 32) {
        if (k0 + 64 < K) {
            __builtin_prefetch(Arow0 + k0 + 64, 0, 1);
            __builtin_prefetch(Arow1 + k0 + 64, 0, 1);
            __builtin_prefetch(Brow + k0 + 64, 0, 1);
        }
        Frag fa0, fa1;
        fa0.q2[0] = *(const uint4*)(Arow0 + k0);
        fa0.q2[1] = *(const uint4*)(Arow0 + k0 + 16);
        fa1.q2[0] = *(const uint4*)(Arow1 + k0);
        fa1.q2[1] = *(const uint4*)(Arow1 + k0 + 16);
#pragma unroll
        for (int j = 0; j < 4; ++j) {
            const unsigned short* bp_ = Brow + (size_t)j * 16 * K + k0;
            Frag fb;
            fb.q2[0] = *(const uint4*)(bp_);
            fb.q2[1] = *(const uint4*)(bp_ + 16);
            acc[0][j] = __builtin_amdgcn_wmma_f32_16x16x32_bf16(
                false, fa0.v, false, fb.v, (short)0, acc[0][j], false, false);
            acc[1][j] = __builtin_amdgcn_wmma_f32_16x16x32_bf16(
                false, fa1.v, false, fb.v, (short)0, acc[1][j], false, false);
        }
    }

    int lh8 = lh * 8;
#pragma unroll
    for (int mt = 0; mt < 2; ++mt) {
#pragma unroll
        for (int j = 0; j < 4; ++j) {
            int n = n0 + j * 16 + lm;
            float bv = bias ? bias[n] : 0.f;
#pragma unroll
            for (int r = 0; r < 8; ++r) {
                int m = m0 + mt * 16 + lh8 + r;
                float val = acc[mt][j][r] + bv;
                if (act == 1) val = gelu_f(val);
                if (resid) val += resid[(size_t)m * N + n];
                if (Cb) Cb[(size_t)m * N + n] = f2bf(val);
                else    C [(size_t)m * N + n] = val;
            }
        }
    }
}

// ---------------------------------------------------------------- autocorr
// corr[b,tau] += (1/D) * circular-diagonal sums of Q_b * K_b^T via WMMA,
// accumulated into an LDS histogram per block, flushed with global atomics.
__global__ void autocorr_kernel(const unsigned short* __restrict__ Qb,
                                const unsigned short* __restrict__ Kb,
                                float* __restrict__ corr) {
    __shared__ float lc[SEQQ];
    for (int i = threadIdx.x; i < SEQQ; i += blockDim.x) lc[i] = 0.f;
    __syncthreads();

    const int TJ = SEQQ / 16;            // 160
    int b  = blockIdx.x / TJ;
    int tj = blockIdx.x % TJ;
    int wave = threadIdx.x >> 5, lane = threadIdx.x & 31;
    int lm = lane & 15, lh = lane >> 4;
    int j0 = tj << 4;

    const unsigned short* Qbase = Qb + (size_t)b * SEQQ * DMODEL;
    const unsigned short* Brow  = Kb + (size_t)b * SEQQ * DMODEL
                                     + (size_t)(j0 + lm) * DMODEL + lh * 8;

    for (int ti = wave; ti < TJ; ti += 8) {
        int i0 = ti << 4;
        const unsigned short* Arow = Qbase + (size_t)(i0 + lm) * DMODEL + lh * 8;
        v8f acc = {};
        for (int k0 = 0; k0 < DMODEL; k0 += 32) {
            Frag fa, fb;
            fa.q2[0] = *(const uint4*)(Arow + k0);
            fa.q2[1] = *(const uint4*)(Arow + k0 + 16);
            fb.q2[0] = *(const uint4*)(Brow + k0);
            fb.q2[1] = *(const uint4*)(Brow + k0 + 16);
            acc = __builtin_amdgcn_wmma_f32_16x16x32_bf16(
                false, fa.v, false, fb.v, (short)0, acc, false, false);
        }
        int mb = i0 + lh * 8, j = j0 + lm;
#pragma unroll
        for (int r = 0; r < 8; ++r) {
            int tau = mb + r - j;
            if (tau < 0) tau += SEQQ;
            atomicAdd(&lc[tau], acc[r]);
        }
    }
    __syncthreads();
    for (int t = threadIdx.x; t < SEQQ; t += blockDim.x)
        atomicAdd(&corr[(size_t)b * SEQQ + t], lc[t] * (1.0f / (float)DMODEL));
}

// ---------------------------------------------------------------- top-k + softmax
__global__ void topk_kernel(const float* __restrict__ corr,
                            int* __restrict__ dly, float* __restrict__ wts) {
    int b = blockIdx.x;
    const float* c = corr + (size_t)b * SEQQ;
    __shared__ float sv[256];
    __shared__ int   si[256];
    __shared__ int   chosen[TOPK];
    __shared__ float chval[TOPK];
    for (int k = 0; k < TOPK; ++k) {
        float best = -3.4e38f; int bi = SEQQ;
        for (int t = threadIdx.x; t < SEQQ; t += 256) {
            bool skip = false;
            for (int p = 0; p < k; ++p) if (chosen[p] == t) skip = true;
            float v = c[t];
            if (!skip && (v > best || (v == best && t < bi))) { best = v; bi = t; }
        }
        sv[threadIdx.x] = best; si[threadIdx.x] = bi;
        __syncthreads();
        for (int s = 128; s > 0; s >>= 1) {
            if (threadIdx.x < s) {
                float ov = sv[threadIdx.x + s]; int oi = si[threadIdx.x + s];
                if (ov > sv[threadIdx.x] ||
                    (ov == sv[threadIdx.x] && oi < si[threadIdx.x])) {
                    sv[threadIdx.x] = ov; si[threadIdx.x] = oi;
                }
            }
            __syncthreads();
        }
        if (threadIdx.x == 0) { chosen[k] = si[0]; chval[k] = sv[0]; }
        __syncthreads();
    }
    if (threadIdx.x == 0) {
        float mx = chval[0];
        for (int k = 1; k < TOPK; ++k) mx = fmaxf(mx, chval[k]);
        float e[TOPK], s = 0.f;
        for (int k = 0; k < TOPK; ++k) { e[k] = __expf(chval[k] - mx); s += e[k]; }
        for (int k = 0; k < TOPK; ++k) {
            wts[b * TOPK + k] = e[k] / s;
            dly[b * TOPK + k] = chosen[k];
        }
    }
}

// ---------------------------------------------------------------- delay aggregation
// att (bf16, ready as GEMM A operand) = sum_k w_k * roll(v, -delay_k)
__global__ void agg_kernel(const float* __restrict__ v,
                           const int* __restrict__ dly,
                           const float* __restrict__ wts,
                           unsigned short* __restrict__ att_bf) {
    size_t idx = (size_t)blockIdx.x * blockDim.x + threadIdx.x;
    if (idx >= (size_t)BATCH * SEQQ * DMODEL) return;
    int d = (int)(idx % DMODEL);
    size_t bs = idx / DMODEL;
    int s = (int)(bs % SEQQ);
    int b = (int)(bs / SEQQ);
    float acc = 0.f;
#pragma unroll
    for (int k = 0; k < TOPK; ++k) {
        int sp = s + dly[b * TOPK + k];
        if (sp >= SEQQ) sp -= SEQQ;
        acc += wts[b * TOPK + k] * v[((size_t)b * SEQQ + sp) * DMODEL + d];
    }
    att_bf[idx] = f2bf(acc);
}

// ---------------------------------------------------------------- moving-avg subtract
__global__ void ma_sub_kernel(const float* __restrict__ x, float* __restrict__ y) {
    size_t idx = (size_t)blockIdx.x * blockDim.x + threadIdx.x;
    if (idx >= (size_t)BATCH * SEQQ * DMODEL) return;
    int d = (int)(idx % DMODEL);
    size_t bs = idx / DMODEL;
    int s = (int)(bs % SEQQ);
    int b = (int)(bs / SEQQ);
    const float* base = x + (size_t)b * SEQQ * DMODEL + d;
    float acc = 0.f;
#pragma unroll
    for (int j = -12; j <= 12; ++j) {
        int ss = s + j;
        ss = ss < 0 ? 0 : (ss >= SEQQ ? SEQQ - 1 : ss);
        acc += base[(size_t)ss * DMODEL];
    }
    y[idx] = x[idx] - acc * (1.0f / 25.0f);
}

// ---------------------------------------------------------------- layernorm (per token)
__global__ void ln_kernel(const float* __restrict__ h,
                          const float* __restrict__ gamma,
                          const float* __restrict__ beta,
                          float* __restrict__ out) {
    int tok = blockIdx.x;
    const float* x = h + (size_t)tok * DMODEL;
    __shared__ float red[256];
    __shared__ float bc[2];
    int tid = threadIdx.x;
    float a = x[tid], b2 = x[tid + 256];
    red[tid] = a + b2;
    __syncthreads();
    for (int s = 128; s > 0; s >>= 1) {
        if (tid < s) red[tid] += red[tid + s];
        __syncthreads();
    }
    if (tid == 0) bc[0] = red[0] / (float)DMODEL;
    __syncthreads();
    float mu = bc[0];
    float da = a - mu, db = b2 - mu;
    red[tid] = da * da + db * db;
    __syncthreads();
    for (int s = 128; s > 0; s >>= 1) {
        if (tid < s) red[tid] += red[tid + s];
        __syncthreads();
    }
    if (tid == 0) bc[1] = red[0] / (float)DMODEL;
    __syncthreads();
    float inv = rsqrtf(bc[1] + 1e-5f);
    out[(size_t)tok * DMODEL + tid]       = da * inv * gamma[tid] + beta[tid];
    out[(size_t)tok * DMODEL + tid + 256] = db * inv * gamma[tid + 256] + beta[tid + 256];
}

// mean over sequence axis per (b,d)
__global__ void colmean_kernel(const float* __restrict__ xh, float* __restrict__ cm) {
    int idx = blockIdx.x * blockDim.x + threadIdx.x;
    if (idx >= BATCH * DMODEL) return;
    int b = idx / DMODEL, d = idx % DMODEL;
    float s = 0.f;
    const float* p = xh + (size_t)b * SEQQ * DMODEL + d;
    for (int t = 0; t < SEQQ; ++t) s += p[(size_t)t * DMODEL];
    cm[idx] = s / (float)SEQQ;
}

// out[b] = sum_{s,d} gelu(xh - colmean) * Wp[s*D+d] + bp
__global__ void final_kernel(const float* __restrict__ xh,
                             const float* __restrict__ cm,
                             const float* __restrict__ Wp,
                             const float* __restrict__ bp,
                             float* __restrict__ out) {
    int b = blockIdx.x;
    __shared__ float red[256];
    float acc = 0.f;
    const float* base = xh + (size_t)b * SEQQ * DMODEL;
    for (size_t i = threadIdx.x; i < (size_t)SEQQ * DMODEL; i += 256) {
        int d = (int)(i % DMODEL);
        float v = gelu_f(base[i] - cm[b * DMODEL + d]);
        acc += v * Wp[i];
    }
    red[threadIdx.x] = acc;
    __syncthreads();
    for (int s = 128; s > 0; s >>= 1) {
        if (threadIdx.x < s) red[threadIdx.x] += red[threadIdx.x + s];
        __syncthreads();
    }
    if (threadIdx.x == 0) out[b] = red[0] + bp[0];
}

// ---------------------------------------------------------------- host
extern "C" void kernel_launch(void* const* d_in, const int* in_sizes, int n_in,
                              void* d_out, int out_size, void* d_ws, size_t ws_size,
                              hipStream_t stream) {
    (void)in_sizes; (void)n_in; (void)out_size; (void)ws_size;
    const float* data  = (const float*)d_in[0];
    const float* mask  = (const float*)d_in[1];
    const float* Wemb  = (const float*)d_in[2];
    const float* Wq    = (const float*)d_in[3];
    const float* bq    = (const float*)d_in[4];
    const float* Wk    = (const float*)d_in[5];
    const float* bk    = (const float*)d_in[6];
    const float* Wv    = (const float*)d_in[7];
    const float* bv    = (const float*)d_in[8];
    const float* Wo    = (const float*)d_in[9];
    const float* bo    = (const float*)d_in[10];
    const float* Wc1   = (const float*)d_in[11];
    const float* Wc2   = (const float*)d_in[12];
    const float* gamma = (const float*)d_in[13];
    const float* beta  = (const float*)d_in[14];
    const float* Wp    = (const float*)d_in[15];
    const float* bp    = (const float*)d_in[16];
    float* outp = (float*)d_out;

    const size_t HSZ   = (size_t)BATCH * SEQQ * DMODEL;   // 20,971,520
    const size_t MIDSZ = (size_t)BATCH * SEQQ * DFF;      // 83,886,080

    float* h    = (float*)d_ws;
    float* q    = h + HSZ;
    float* k    = q + HSZ;
    float* v    = k + HSZ;
    float* corr = v + HSZ;                     // BATCH*SEQQ
    float* wts  = corr + (size_t)BATCH * SEQQ; // BATCH*TOPK
    int*   dly  = (int*)(wts + BATCH * TOPK);
    float* cm   = (float*)(dly + BATCH * TOPK); // BATCH*DMODEL
    unsigned short* xb = (unsigned short*)(cm + BATCH * DMODEL); // MIDSZ bf16
    unsigned short* hb = xb + MIDSZ;           // HSZ bf16 (A-operand staging)
    unsigned short* wb = hb + HSZ;             // up to 1,048,576 bf16 weights

    const int T = 256;
    const int gH = (int)(HSZ / T);             // 81920
    dim3 blk(T);

    // GEMM grids: (M/32 row-groups) * (N/64 col-groups) waves, 8 waves/block
    const int gB512  = (BATCH * SEQQ / 32) * (DMODEL / 64) / 8;  // 1280
    const int gB2048 = (BATCH * SEQQ / 32) * (DFF / 64) / 8;     // 5120

    embed_kernel<<<gH, blk, 0, stream>>>(data, mask, Wemb, h);

    for (int l = 0; l < ELAYERS; ++l) {
        const float* Wq_l = Wq + (size_t)l * DMODEL * DMODEL;
        const float* Wk_l = Wk + (size_t)l * DMODEL * DMODEL;
        const float* Wv_l = Wv + (size_t)l * DMODEL * DMODEL;
        const float* Wo_l = Wo + (size_t)l * DMODEL * DMODEL;
        const float* Wc1_l = Wc1 + (size_t)l * DMODEL * DFF;
        const float* Wc2_l = Wc2 + (size_t)l * DFF * DMODEL;
        const float* bq_l = bq + l * DMODEL;
        const float* bk_l = bk + l * DMODEL;
        const float* bv_l = bv + l * DMODEL;
        const float* bo_l = bo + l * DMODEL;

        // --- QKV projections: q,k straight to bf16 staging; v to fp32 ---
        f2bf_kernel<<<gH, blk, 0, stream>>>(h, hb, HSZ);
        wtT_kernel<<<(DMODEL * DMODEL) / T, blk, 0, stream>>>(Wq_l, wb, DMODEL, DMODEL);
        wmma_gemm_kernel<<<gB512, blk, 0, stream>>>(hb, wb, bq_l, nullptr,
                                                    nullptr, xb,
                                                    BATCH * SEQQ, DMODEL, DMODEL, 0);
        wtT_kernel<<<(DMODEL * DMODEL) / T, blk, 0, stream>>>(Wk_l, wb, DMODEL, DMODEL);
        wmma_gemm_kernel<<<gB512, blk, 0, stream>>>(hb, wb, bk_l, nullptr,
                                                    nullptr, xb + HSZ,
                                                    BATCH * SEQQ, DMODEL, DMODEL, 0);
        wtT_kernel<<<(DMODEL * DMODEL) / T, blk, 0, stream>>>(Wv_l, wb, DMODEL, DMODEL);
        wmma_gemm_kernel<<<gB512, blk, 0, stream>>>(hb, wb, bv_l, nullptr,
                                                    v, nullptr,
                                                    BATCH * SEQQ, DMODEL, DMODEL, 0);

        // --- autocorrelation via WMMA Q.K^T circular diagonals ---
        hipMemsetAsync(corr, 0, (size_t)BATCH * SEQQ * sizeof(float), stream);
        autocorr_kernel<<<BATCH * (SEQQ / 16), blk, 0, stream>>>(xb, xb + HSZ, corr);
        topk_kernel<<<BATCH, blk, 0, stream>>>(corr, dly, wts);
        agg_kernel<<<gH, blk, 0, stream>>>(v, dly, wts, xb);   // att (bf16) -> xb

        // --- out projection + residual, then x = x - MA(x) ---
        wtT_kernel<<<(DMODEL * DMODEL) / T, blk, 0, stream>>>(Wo_l, wb, DMODEL, DMODEL);
        wmma_gemm_kernel<<<gB512, blk, 0, stream>>>(xb, wb, bo_l, h,
                                                    k, nullptr,
                                                    BATCH * SEQQ, DMODEL, DMODEL, 0);
        ma_sub_kernel<<<gH, blk, 0, stream>>>(k, h);

        // --- FFN: mid = gelu(x@Wc1) as bf16 directly; z = x + mid@Wc2; z -= MA(z) ---
        f2bf_kernel<<<gH, blk, 0, stream>>>(h, hb, HSZ);
        wtT_kernel<<<(DMODEL * DFF) / T, blk, 0, stream>>>(Wc1_l, wb, DMODEL, DFF);
        wmma_gemm_kernel<<<gB2048, blk, 0, stream>>>(hb, wb, nullptr, nullptr,
                                                     nullptr, xb,
                                                     BATCH * SEQQ, DFF, DMODEL, 1);
        wtT_kernel<<<(DFF * DMODEL) / T, blk, 0, stream>>>(Wc2_l, wb, DFF, DMODEL);
        wmma_gemm_kernel<<<gB512, blk, 0, stream>>>(xb, wb, nullptr, h,
                                                    k, nullptr,
                                                    BATCH * SEQQ, DMODEL, DFF, 0);
        ma_sub_kernel<<<gH, blk, 0, stream>>>(k, h);
    }

    // --- final layernorm (+seq-mean subtract) -> gelu -> Wp projection ---
    ln_kernel<<<BATCH * SEQQ, blk, 0, stream>>>(h, gamma, beta, q);
    colmean_kernel<<<(BATCH * DMODEL) / T, blk, 0, stream>>>(q, cm);
    final_kernel<<<BATCH, blk, 0, stream>>>(q, cm, Wp, bp, outp);
}
